// Downsample_PASA_group_softmax_17652315587405
// MI455X (gfx1250) — compile-verified, matrix operand order checked by reading, not run on hardware
//
#include <hip/hip_runtime.h>
#include <hip/hip_bf16.h>

// ---------------------------------------------------------------------------
// Downsample_PASA_group_softmax for MI455X (gfx1250, wave32, WMMA).
//
//   k_prep  : zero BN accumulators; build per-lane WMMA B-fragment images of
//             the conv weights (bf16 pairs, K reordered as k'=(di*3+dj)*64+c,
//             N padded 18->32 with zeros) into workspace.
//   k_conv  : 3x3 reflect conv -> sigma(18ch) as bf16 WMMA GEMM.
//             Block = one (n,h) row (256 pixels), 8 waves x 32 pixels.
//             Per 32-K chunk: 2x ds_load_b128 per A tile, 4x for B,
//             4x v_wmma_f32_16x16x32_bf16 (A reused across both N tiles,
//             B reused across both M tiles). Fused per-channel sum/sumsq
//             (BN batch stats); sigma stored only at stride-2 positions.
//   k_stats : finalize BN scale/shift.
//   k_agg   : BN -> softmax(18) -> grouped 3x3 adaptive aggregation.
// ---------------------------------------------------------------------------

typedef __bf16  bf16_t;
typedef bf16_t  v16bf __attribute__((ext_vector_type(16)));
typedef float   v8f   __attribute__((ext_vector_type(8)));

#define NB    4
#define CH    64
#define HH    256
#define WW    256
#define MCH   18          // G*K*K output channels
#define HO    128
#define WO    128
#define CSTR  72          // c-dim stride (u16) in x LDS tile (64 used, padded)
#define XCOLS 258         // 256 pixels + 2 halo
#define XS32  ((XCOLS * 3 * CSTR) / 2)   // u32 words in x tile = 27864
#define WBN   (18 * 2 * 32 * 8)          // u32 words of B-fragment image = 9216

__device__ __forceinline__ int reflect_i(int i, int n) {
    return i < 0 ? -i : (i >= n ? 2 * n - 2 - i : i);
}

__device__ __forceinline__ unsigned f2bf(float f) {
    unsigned u = __float_as_uint(f);
    unsigned r = u + 0x7FFFu + ((u >> 16) & 1u);   // round-to-nearest-even
    return r >> 16;
}

// ---------------------------------------------------------------------------
// Prep: zero stats + build B-fragment image gwb[((kc*2+mt)*32+lane)*8+v].
// ---------------------------------------------------------------------------
__global__ void k_prep(const float* __restrict__ w, unsigned* __restrict__ gwb,
                       float* __restrict__ stats) {
    int gid = blockIdx.x * 256 + threadIdx.x;
    if (blockIdx.x == 0 && threadIdx.x < 128) stats[threadIdx.x] = 0.0f;
    if (gid < WBN) {
        int v    = gid & 7;
        int lane = (gid >> 3) & 31;
        int mt   = (gid >> 8) & 1;
        int kc   = gid >> 9;
        int m    = mt * 16 + (lane & 15);
        int k0   = kc * 32 + (lane >> 4) * 16 + 2 * v;   // reordered K index
        unsigned lo = 0, hi = 0;
        if (m < MCH) {
            int q0 = k0 >> 6, c0 = k0 & 63;              // k' = q*64 + c
            int q1 = (k0 + 1) >> 6, c1 = (k0 + 1) & 63;
            lo = f2bf(w[m * 576 + c0 * 9 + q0]);         // w[m][c][di][dj], q=di*3+dj
            hi = f2bf(w[m * 576 + c1 * 9 + q1]);
        }
        gwb[gid] = lo | (hi << 16);
    }
}

// ---------------------------------------------------------------------------
// Conv + BN-stat kernel. Grid: 1024 blocks (n x h), 256 threads = 8 waves,
// each wave owns 32 consecutive pixels (two 16-row M tiles).
// ---------------------------------------------------------------------------
__global__ __launch_bounds__(256) void k_conv(const float* __restrict__ x,
                                              const unsigned* __restrict__ gwb,
                                              float* __restrict__ sigma_s,
                                              float* __restrict__ gsum,
                                              float* __restrict__ gsumsq) {
    __shared__ __align__(16) unsigned xs32[XS32];   // bf16 x tile [(col*3+di)*72 + c]
    __shared__ __align__(16) unsigned lwb[WBN];     // B fragment image
    __shared__ float s_sum[32], s_sq[32];

    const int tid = threadIdx.x;
    const int n   = blockIdx.x >> 8;
    const int h   = blockIdx.x & 255;

    if (tid < 32) { s_sum[tid] = 0.0f; s_sq[tid] = 0.0f; }

    // Weights (B fragments) -> LDS, coalesced 16B copies.
    {
        const uint4* g4 = (const uint4*)gwb;
        uint4* l4 = (uint4*)lwb;
        for (int i = tid; i < WBN / 4; i += 256) l4[i] = g4[i];
    }
    // x (reflect-padded 3-row window) -> LDS as packed bf16.
    unsigned short* xsh = (unsigned short*)xs32;
    const float* xn = x + (size_t)n * CH * HH * WW;
    for (int idx = tid; idx < 192 * XCOLS; idx += 256) {
        int rowc = idx / XCOLS;                 // c*3 + di
        int col  = idx - rowc * XCOLS;
        int c = rowc / 3, di = rowc - c * 3;
        int hs  = reflect_i(h + di - 1, HH);
        int ws2 = reflect_i(col - 1, WW);
        xsh[(col * 3 + di) * CSTR + c] = (unsigned short)f2bf(xn[((size_t)c * HH + hs) * WW + ws2]);
    }
    __syncthreads();

    const int lane  = tid & 31;
    const int wv    = tid >> 5;
    const int r     = lane & 15;
    const int khalf = lane >> 4;
    const int tl0   = wv * 32 + r;        // M tile 0 row (pixel)
    const int tl1   = tl0 + 16;           // M tile 1 row
    const int cbh   = khalf * 8;          // u32 c-offset from K half
    const int m0    = r;                  // C column (output channel) of this lane

    v8f acc00 = {}, acc01 = {}, acc10 = {}, acc11 = {};
    union Frag { v16bf v; unsigned u[8]; };
    Frag a0, a1, b0, b1;

    #pragma unroll
    for (int kc = 0; kc < 18; ++kc) {
        const int q  = kc >> 1;
        const int di = q / 3, dj = q - 3 * di;       // compile-time after unroll
        const int cb = (kc & 1) * 16 + cbh;          // u32 c-offset
        #pragma unroll
        for (int v = 0; v < 8; ++v) {
            a0.u[v] = xs32[((tl0 + dj) * 3 + di) * (CSTR / 2) + cb + v];
            a1.u[v] = xs32[((tl1 + dj) * 3 + di) * (CSTR / 2) + cb + v];
            b0.u[v] = lwb[kc * 512 + lane * 8 + v];
            b1.u[v] = lwb[kc * 512 + 256 + lane * 8 + v];
        }
        acc00 = __builtin_amdgcn_wmma_f32_16x16x32_bf16(false, a0.v, false, b0.v,
                                                        (short)0, acc00, false, false);
        acc01 = __builtin_amdgcn_wmma_f32_16x16x32_bf16(false, a0.v, false, b1.v,
                                                        (short)0, acc01, false, false);
        acc10 = __builtin_amdgcn_wmma_f32_16x16x32_bf16(false, a1.v, false, b0.v,
                                                        (short)0, acc10, false, false);
        acc11 = __builtin_amdgcn_wmma_f32_16x16x32_bf16(false, a1.v, false, b1.v,
                                                        (short)0, acc11, false, false);
    }

    // BN batch statistics (lane holds 16 pixels of channel m0 and of m0+16).
    float s0 = 0.f, q0 = 0.f, s1 = 0.f, q1 = 0.f;
    #pragma unroll
    for (int v = 0; v < 8; ++v) {
        float f;
        f = acc00[v]; s0 += f; q0 += f * f;
        f = acc10[v]; s0 += f; q0 += f * f;
        f = acc01[v]; s1 += f; q1 += f * f;
        f = acc11[v]; s1 += f; q1 += f * f;
    }
    atomicAdd(&s_sum[m0], s0);
    atomicAdd(&s_sq[m0],  q0);
    if (m0 + 16 < MCH) { atomicAdd(&s_sum[m0 + 16], s1); atomicAdd(&s_sq[m0 + 16], q1); }

    // Store sigma only at stride-2 positions (C row = v + 8*khalf).
    if ((h & 1) == 0) {
        const int ho = h >> 1;
        #pragma unroll
        for (int v = 0; v < 8; ++v) {
            int pix0 = wv * 32 + v + 8 * khalf;
            if ((pix0 & 1) == 0) {
                int wo = pix0 >> 1;
                sigma_s[(((size_t)n * MCH + m0) * HO + ho) * WO + wo] = acc00[v];
                if (m0 + 16 < MCH)
                    sigma_s[(((size_t)n * MCH + m0 + 16) * HO + ho) * WO + wo] = acc01[v];
            }
            int pix1 = pix0 + 16;
            if ((pix1 & 1) == 0) {
                int wo = pix1 >> 1;
                sigma_s[(((size_t)n * MCH + m0) * HO + ho) * WO + wo] = acc10[v];
                if (m0 + 16 < MCH)
                    sigma_s[(((size_t)n * MCH + m0 + 16) * HO + ho) * WO + wo] = acc11[v];
            }
        }
    }
    __syncthreads();
    if (tid < MCH) {
        atomicAdd(&gsum[tid],   s_sum[tid]);
        atomicAdd(&gsumsq[tid], s_sq[tid]);
    }
}

__global__ void k_stats(const float* __restrict__ gsum,
                        const float* __restrict__ gsumsq,
                        const float* __restrict__ gamma,
                        const float* __restrict__ beta,
                        float* __restrict__ scale,
                        float* __restrict__ shift) {
    int m = threadIdx.x;
    if (m < MCH) {
        const float inv_n = 1.0f / (float)(NB * HH * WW);
        float mean = gsum[m] * inv_n;
        float var  = gsumsq[m] * inv_n - mean * mean;
        float sc   = gamma[m] * rsqrtf(var + 1e-5f);
        scale[m] = sc;
        shift[m] = beta[m] - mean * sc;
    }
}

// ---------------------------------------------------------------------------
// BN -> softmax(18) -> grouped 3x3 adaptive aggregation at stride-2 points.
// ---------------------------------------------------------------------------
__global__ __launch_bounds__(256) void k_agg(const float* __restrict__ x,
                                             const float* __restrict__ sigma_s,
                                             const float* __restrict__ scale,
                                             const float* __restrict__ shift,
                                             float* __restrict__ out) {
    __shared__ float sm[MCH][WO];
    const int tid = threadIdx.x;
    const int n   = blockIdx.x >> 7;
    const int ho  = blockIdx.x & 127;

    if (tid < WO) {
        const int wo = tid;
        float vals[MCH];
        float mx = -3.4e38f;
        #pragma unroll
        for (int m = 0; m < MCH; ++m) {
            float s = sigma_s[(((size_t)n * MCH + m) * HO + ho) * WO + wo];
            s = s * scale[m] + shift[m];
            vals[m] = s;
            mx = fmaxf(mx, s);
        }
        float sum = 0.f;
        #pragma unroll
        for (int m = 0; m < MCH; ++m) { float e = __expf(vals[m] - mx); vals[m] = e; sum += e; }
        float inv = 1.0f / sum;
        #pragma unroll
        for (int m = 0; m < MCH; ++m) sm[m][wo] = vals[m] * inv;
    }
    __syncthreads();

    const int wo   = tid & 127;
    const int coff = tid >> 7;
    const float* xn = x + (size_t)n * CH * HH * WW;
    const int h0 = 2 * ho, w0 = 2 * wo;
    int rr[3], cc[3];
    #pragma unroll
    for (int i = 0; i < 3; ++i) {
        rr[i] = reflect_i(h0 + i - 1, HH);
        cc[i] = reflect_i(w0 + i - 1, WW);
    }
    for (int it = 0; it < 32; ++it) {
        const int c = it * 2 + coff;
        const float* xc = xn + (size_t)c * HH * WW;
        const int g = c >> 5;
        float accv = 0.f;
        #pragma unroll
        for (int i = 0; i < 3; ++i)
            #pragma unroll
            for (int j = 0; j < 3; ++j)
                accv += xc[rr[i] * WW + cc[j]] * sm[g * 9 + i * 3 + j][wo];
        out[(((size_t)n * CH + c) * HO + ho) * WO + wo] = accv;
    }
}

extern "C" void kernel_launch(void* const* d_in, const int* in_sizes, int n_in,
                              void* d_out, int out_size, void* d_ws, size_t ws_size,
                              hipStream_t stream) {
    const float* x      = (const float*)d_in[0];   // (4,64,256,256)
    const float* conv_w = (const float*)d_in[1];   // (18,64,3,3)
    const float* gamma  = (const float*)d_in[2];   // (18,)
    const float* beta   = (const float*)d_in[3];   // (18,)
    float* out = (float*)d_out;                    // (4,64,128,128)

    float* ws       = (float*)d_ws;
    float* gsum     = ws;                          // 18 (padded 32)
    float* gsumsq   = ws + 32;                     // 18 (padded 32)
    float* scale    = ws + 64;                     // 18 (padded 32)
    float* shift    = ws + 96;                     // 18 (padded 32)
    unsigned* gwb   = (unsigned*)(ws + 128);       // 9216 u32 B-fragment image
    float* sigma_s  = ws + 128 + WBN;              // 4*18*128*128 floats

    k_prep <<<36, 256, 0, stream>>>(conv_w, gwb, ws);
    k_conv <<<NB * HH, 256, 0, stream>>>(x, gwb, sigma_s, gsum, gsumsq);
    k_stats<<<1, 32, 0, stream>>>(gsum, gsumsq, gamma, beta, scale, shift);
    k_agg  <<<NB * HO, 256, 0, stream>>>(x, sigma_s, scale, shift, out);
}
